// MultiLayerAttention_11751030522297
// MI455X (gfx1250) — matrix operand order repaired and checked by reading, not compiled, not run on hardware
//
#include <hip/hip_runtime.h>
#include <cmath>

// ---------------------------------------------------------------------------
// MultiLayerAttention for MI455X (gfx1250): f16 WMMA everywhere, flash-style
// online-softmax attention, all activations staged in f16 scratch.
// ---------------------------------------------------------------------------

typedef __attribute__((ext_vector_type(16))) _Float16 v16h;
typedef __attribute__((ext_vector_type(8)))  _Float16 v8h;
typedef __attribute__((ext_vector_type(8)))  float    v8f;

#define HEADS 4
#define BATCH 4

static __device__ __forceinline__ v16h cat8(v8h a, v8h b) {
  return __builtin_shufflevector(a, b, 0,1,2,3,4,5,6,7,8,9,10,11,12,13,14,15);
}

// ---------------------------------------------------------------------------
// LayerNorm over channel dim of NCHW input -> f16 row-major [B*N, C]
// x: [B, C, N] (N = H*W). Threads over (b, n): channel reads are coalesced.
// ---------------------------------------------------------------------------
__global__ void ln_k(const float* __restrict__ x, const float* __restrict__ g,
                     const float* __restrict__ beta, _Float16* __restrict__ out,
                     int C, int N) {
  int idx = blockIdx.x * blockDim.x + threadIdx.x;
  int total = BATCH * N;
  if (idx >= total) return;
  int b = idx / N, n = idx - b * N;
  const float* xp = x + (size_t)b * C * N + n;
  float s = 0.f, s2 = 0.f;
  for (int c = 0; c < C; ++c) { float v = xp[(size_t)c * N]; s += v; s2 += v * v; }
  float mean = s / C;
  float var  = s2 / C - mean * mean;
  float inv  = rsqrtf(var + 1e-5f);
  _Float16* op = out + (size_t)idx * C;
  for (int c = 0; c < C; ++c) {
    float v = (xp[(size_t)c * N] - mean) * inv * g[c] + beta[c];
    op[c] = (_Float16)v;
  }
}

// ---------------------------------------------------------------------------
// Weight convert+transpose: f32 [K, Nc] -> f16 [Nc, K]  (WMMA-B friendly)
// ---------------------------------------------------------------------------
__global__ void wt_k(const float* __restrict__ w, _Float16* __restrict__ wt,
                     int K, int Nc) {
  int idx = blockIdx.x * blockDim.x + threadIdx.x;
  if (idx >= K * Nc) return;
  int k = idx / Nc, n = idx - k * Nc;
  wt[(size_t)n * K + k] = (_Float16)w[idx];
}

// ---------------------------------------------------------------------------
// Generic WMMA GEMM: C[M, NT*16-wide tiles] = A[M,K] (f16 rm) @ Bt (f16 [Nc,K]).
// One wave computes a 16 x (NT*16) tile; K-loop in steps of 32; branch-free.
// MODE 0: store f16 row-major to o16 (ld = Ncols)
// MODE 1: cols [0,inner) -> o16 row-major [M,inner] (K matrix);
//         cols [inner,2*inner) -> vt transposed [b][dim][key] (V matrix)
// MODE 2: store f32, NCHW scatter: o32[((b*Cc + col)*Ntok) + tok]
// ---------------------------------------------------------------------------
template <int NT, int MODE>
__global__ __launch_bounds__(32)
void gemm_k(const _Float16* __restrict__ A, const _Float16* __restrict__ Bt,
            int K, int Ncols,
            _Float16* __restrict__ o16, _Float16* __restrict__ vt,
            float* __restrict__ o32, int inner, int Nj, int Ntok, int Cc) {
  const int lane = threadIdx.x;
  const int hi   = lane >> 4;     // which half-wave
  const int l16  = lane & 15;
  const int m0   = blockIdx.x * 16;
  const int n0   = blockIdx.y * (NT * 16);

  v8f acc[NT] = {};
  const _Float16* arow = A + (size_t)(m0 + l16) * K;
  const _Float16* brow = Bt + (size_t)(n0 + l16) * K + hi * 16;

  for (int kk = 0; kk < K; kk += 32) {
    // A fragment (16x32 f16): lanes 0-15 rows M=0-15 K{0..15},
    // lanes 16-31 same rows, K{16..31}
    v8h a0 = *(const v8h*)(arow + kk + hi * 16);
    v8h a1 = *(const v8h*)(arow + kk + hi * 16 + 8);
    v16h af = cat8(a0, a1);
#pragma unroll
    for (int t = 0; t < NT; ++t) {
      // B fragment (32x16): lane holds col (n0 + t*16 + l16), 16 contiguous K halves
      v16h bf = *(const v16h*)(brow + (size_t)t * 16 * K + kk);
      acc[t] = __builtin_amdgcn_wmma_f32_16x16x32_f16(
          false, af, false, bf, (short)0, acc[t], false, false);
    }
  }

#pragma unroll
  for (int t = 0; t < NT; ++t) {
    int col = n0 + t * 16 + l16;
#pragma unroll
    for (int r = 0; r < 8; ++r) {
      int row = m0 + hi * 8 + r;          // C layout: VGPR r -> M = hi*8 + r
      float v = acc[t][r];
      if (MODE == 0) {
        o16[(size_t)row * Ncols + col] = (_Float16)v;
      } else if (MODE == 1) {
        if (col < inner) {
          o16[(size_t)row * inner + col] = (_Float16)v;          // K row-major
        } else {
          int dim = col - inner;
          int bb  = row / Nj;
          int key = row - bb * Nj;
          vt[((size_t)bb * inner + dim) * Nj + key] = (_Float16)v; // V transposed
        }
      } else {
        int bb = row / Ntok;
        int n  = row - bb * Ntok;
        o32[((size_t)bb * Cc + col) * Ntok + n] = v;             // NCHW scatter
      }
    }
  }
}

// ---------------------------------------------------------------------------
// Streaming attention: one wave handles 16 query rows for one (batch, head).
// Q: f16 [B*Ni, 4D] row-major; Kb: f16 [B*Nj, 4D] row-major;
// Vt: f16 [B][4D][Nj] (dim-major); O: f16 [B*Ni, 8D], col = h*2D + br*D + dim.
// ---------------------------------------------------------------------------
template <int D>
__global__ __launch_bounds__(32)
void attn_k(const _Float16* __restrict__ Q, const _Float16* __restrict__ Kb,
            const _Float16* __restrict__ Vt, _Float16* __restrict__ O,
            int Ni, int Nj, float scale, int br) {
  constexpr int NK    = D / 32;   // K-dim WMMA steps for Q.K^T
  constexpr int NDT   = D / 16;   // output dim tiles for P.V
  constexpr int INNER = 4 * D;
  __shared__ _Float16 lds[16 * 40];   // 16 rows x 32 cols of P, padded

  const int lane = threadIdx.x;
  const int hi   = lane >> 4;
  const int l16  = lane & 15;
  const int h    = blockIdx.y;
  const int b    = blockIdx.z;

  // Cache Q fragments (A matrix) for all K chunks.
  const _Float16* qp = Q + ((size_t)b * Ni + blockIdx.x * 16 + l16) * INNER + h * D;
  v16h qf[NK];
#pragma unroll
  for (int kc = 0; kc < NK; ++kc) {
    v8h a0 = *(const v8h*)(qp + kc * 32 + hi * 16);
    v8h a1 = *(const v8h*)(qp + kc * 32 + hi * 16 + 8);
    qf[kc] = cat8(a0, a1);
  }

  v8f acc[NDT] = {};
  float mrow[8], lrow[8];
#pragma unroll
  for (int r = 0; r < 8; ++r) { mrow[r] = -3.0e38f; lrow[r] = 0.f; }

  const _Float16* kbase = Kb + (size_t)b * Nj * INNER + h * D;
  const _Float16* vbase = Vt + ((size_t)b * INNER + h * D) * Nj;

  for (int t0 = 0; t0 < Nj; t0 += 32) {
    // Scores for 32 keys as two 16x16 tiles.
    v8f s0 = {}, s1 = {};
#pragma unroll
    for (int kc = 0; kc < NK; ++kc) {
      v16h k0 = *(const v16h*)(kbase + (size_t)(t0 + l16) * INNER + kc * 32 + hi * 16);
      s0 = __builtin_amdgcn_wmma_f32_16x16x32_f16(false, qf[kc], false, k0,
                                                  (short)0, s0, false, false);
    }
#pragma unroll
    for (int kc = 0; kc < NK; ++kc) {
      v16h k1 = *(const v16h*)(kbase + (size_t)(t0 + 16 + l16) * INNER + kc * 32 + hi * 16);
      s1 = __builtin_amdgcn_wmma_f32_16x16x32_f16(false, qf[kc], false, k1,
                                                  (short)0, s1, false, false);
    }

    // Online softmax across the 32 new columns (row-wise shfl reductions).
    float fac[8];
#pragma unroll
    for (int r = 0; r < 8; ++r) {
      float a0 = s0[r] * scale;
      float a1 = s1[r] * scale;
      float mx = fmaxf(a0, a1);
#pragma unroll
      for (int off = 1; off < 16; off <<= 1) mx = fmaxf(mx, __shfl_xor(mx, off, 32));
      float mn = fmaxf(mrow[r], mx);
      float f  = __expf(mrow[r] - mn);
      float p0 = __expf(a0 - mn);
      float p1 = __expf(a1 - mn);
      float rs = p0 + p1;
#pragma unroll
      for (int off = 1; off < 16; off <<= 1) rs += __shfl_xor(rs, off, 32);
      lrow[r] = lrow[r] * f + rs;
      mrow[r] = mn;
      fac[r]  = f;
      lds[(hi * 8 + r) * 40 + l16]      = (_Float16)p0;   // P row, cols 0..15
      lds[(hi * 8 + r) * 40 + 16 + l16] = (_Float16)p1;   // P row, cols 16..31
    }
#pragma unroll
    for (int t = 0; t < NDT; ++t)
#pragma unroll
      for (int r = 0; r < 8; ++r) acc[t][r] *= fac[r];

    // Re-fragment P as A matrix (16x32) from LDS.
    v8h p0v = *(const v8h*)(&lds[l16 * 40 + hi * 16]);
    v8h p1v = *(const v8h*)(&lds[l16 * 40 + hi * 16 + 8]);
    v16h pf = cat8(p0v, p1v);

    // O += P @ V  (V fragment contiguous thanks to transposed Vt layout)
#pragma unroll
    for (int t = 0; t < NDT; ++t) {
      v16h vf = *(const v16h*)(vbase + (size_t)(t * 16 + l16) * Nj + t0 + hi * 16);
      acc[t] = __builtin_amdgcn_wmma_f32_16x16x32_f16(false, pf, false, vf,
                                                      (short)0, acc[t], false, false);
    }
  }

  // Normalize and store: col = h*2D + br*D + dim (matches b h n (2d) concat).
  size_t rb = (size_t)b * Ni + blockIdx.x * 16 + hi * 8;
#pragma unroll
  for (int r = 0; r < 8; ++r) {
    float invl = 1.f / lrow[r];
#pragma unroll
    for (int t = 0; t < NDT; ++t) {
      int col = h * (2 * D) + br * D + t * 16 + l16;
      O[(rb + r) * (size_t)(8 * D) + col] = (_Float16)(acc[t][r] * invl);
    }
  }
}

// ---------------------------------------------------------------------------
// Host orchestration
// ---------------------------------------------------------------------------
extern "C" void kernel_launch(void* const* d_in, const int* in_sizes, int n_in,
                              void* d_out, int out_size, void* d_ws, size_t ws_size,
                              hipStream_t stream) {
  (void)n_in; (void)out_size; (void)ws_size;
  static const int Ns[3] = {4096, 1024, 256};
  static const int Cs[3] = {32, 64, 128};
  static const size_t outOff[3] = {0, (size_t)BATCH * 32 * 64 * 64,
                                   (size_t)BATCH * 32 * 64 * 64 + (size_t)BATCH * 64 * 32 * 32};

  const float* X[3];
  const float *qg[3], *qb[3], *wq[3], *wo[3];
  const float *kg[3][2], *kb[3][2], *wkv[3][2];

  // Detect flattening order from in_sizes (insertion vs alphabetical pytree).
  bool xFirst = (in_sizes[0] == BATCH * 32 * 64 * 64);
  bool sortedKeys = xFirst ? (in_sizes[3] != Cs[0]) : true;
  int p = xFirst ? 3 : 0;
  for (int i = 0; i < 3; ++i) {
    if (!sortedKeys) {
      // insertion order: qn_g, qn_b, wq, kvn_g[0..1], kvn_b[0..1], wkv[0..1], wo
      qg[i] = (const float*)d_in[p++]; qb[i] = (const float*)d_in[p++];
      wq[i] = (const float*)d_in[p++];
      kg[i][0] = (const float*)d_in[p++]; kg[i][1] = (const float*)d_in[p++];
      kb[i][0] = (const float*)d_in[p++]; kb[i][1] = (const float*)d_in[p++];
      wkv[i][0] = (const float*)d_in[p++]; wkv[i][1] = (const float*)d_in[p++];
      wo[i] = (const float*)d_in[p++];
    } else {
      // alphabetical: kvn_b, kvn_g, qn_b, qn_g, wkv, wo, wq
      kb[i][0] = (const float*)d_in[p++]; kb[i][1] = (const float*)d_in[p++];
      kg[i][0] = (const float*)d_in[p++]; kg[i][1] = (const float*)d_in[p++];
      qb[i] = (const float*)d_in[p++]; qg[i] = (const float*)d_in[p++];
      wkv[i][0] = (const float*)d_in[p++]; wkv[i][1] = (const float*)d_in[p++];
      wo[i] = (const float*)d_in[p++]; wq[i] = (const float*)d_in[p++];
    }
  }
  if (xFirst) { X[0] = (const float*)d_in[0]; X[1] = (const float*)d_in[1]; X[2] = (const float*)d_in[2]; }
  else        { X[0] = (const float*)d_in[30]; X[1] = (const float*)d_in[31]; X[2] = (const float*)d_in[32]; }

  char*  ws   = (char*)d_ws;
  float* outf = (float*)d_out;

  for (int i = 0; i < 3; ++i) {
    const int Ci = Cs[i], Ni = Ns[i], inner = 4 * Ci;
    int jb[2], nb2 = 0;
    for (int j = 0; j < 3; ++j) if (j != i) jb[nb2++] = j;

    // Bump allocator over d_ws; scratch is reused across levels (stream-ordered).
    size_t off = 0;
    auto alloc = [&](size_t elems) -> _Float16* {
      _Float16* ptr = (_Float16*)(ws + off);
      off += ((elems * 2 + 255) / 256) * 256;
      return ptr;
    };
    _Float16* lnQ = alloc((size_t)BATCH * Ni * Ci);
    _Float16* lnKV[2];
    for (int br = 0; br < 2; ++br) lnKV[br] = alloc((size_t)BATCH * Ns[jb[br]] * Cs[jb[br]]);
    _Float16* wqT = alloc((size_t)inner * Ci);
    _Float16* wkvT[2];
    for (int br = 0; br < 2; ++br) wkvT[br] = alloc((size_t)2 * inner * Cs[jb[br]]);
    _Float16* woT = alloc((size_t)Ci * 2 * inner);
    _Float16* Qb  = alloc((size_t)BATCH * Ni * inner);
    _Float16 *Kbuf[2], *Vtb[2];
    for (int br = 0; br < 2; ++br) {
      Kbuf[br] = alloc((size_t)BATCH * Ns[jb[br]] * inner);
      Vtb[br]  = alloc((size_t)BATCH * Ns[jb[br]] * inner);
    }
    _Float16* Ob = alloc((size_t)BATCH * Ni * 2 * inner);

    // 1) LayerNorms (Q path + both KV branches)
    {
      int tot = BATCH * Ni;
      ln_k<<<dim3((tot + 255) / 256), dim3(256), 0, stream>>>(X[i], qg[i], qb[i], lnQ, Ci, Ni);
    }
    for (int br = 0; br < 2; ++br) {
      int j = jb[br], tot = BATCH * Ns[j];
      ln_k<<<dim3((tot + 255) / 256), dim3(256), 0, stream>>>(X[j], kg[i][br], kb[i][br],
                                                              lnKV[br], Cs[j], Ns[j]);
    }
    // 2) Weight converts (f32 -> f16 transposed)
    { int tot = Ci * inner;       wt_k<<<(tot + 255) / 256, 256, 0, stream>>>(wq[i], wqT, Ci, inner); }
    for (int br = 0; br < 2; ++br) {
      int j = jb[br], tot = Cs[j] * 2 * inner;
      wt_k<<<(tot + 255) / 256, 256, 0, stream>>>(wkv[i][br], wkvT[br], Cs[j], 2 * inner);
    }
    { int tot = 2 * inner * Ci;   wt_k<<<(tot + 255) / 256, 256, 0, stream>>>(wo[i], woT, 2 * inner, Ci); }

    // 3) Q projection (inner is a multiple of 64 -> NT=4)
    gemm_k<4, 0><<<dim3((BATCH * Ni) / 16, inner / 64), 32, 0, stream>>>(
        lnQ, wqT, Ci, inner, Qb, nullptr, nullptr, 0, 0, 0, 0);
    // 4) KV projections (2*inner multiple of 64 -> NT=4; K row-major, V transposed)
    for (int br = 0; br < 2; ++br) {
      int j = jb[br];
      gemm_k<4, 1><<<dim3((BATCH * Ns[j]) / 16, (2 * inner) / 64), 32, 0, stream>>>(
          lnKV[br], wkvT[br], Cs[j], 2 * inner,
          Kbuf[br], Vtb[br], nullptr, inner, Ns[j], 0, 0);
    }
    // 5) Attention per branch
    float scale = 1.0f / sqrtf((float)Ci);
    for (int br = 0; br < 2; ++br) {
      int j = jb[br];
      dim3 g(Ni / 16, HEADS, BATCH);
      if (Ci == 32)
        attn_k<32><<<g, 32, 0, stream>>>(Qb, Kbuf[br], Vtb[br], Ob, Ni, Ns[j], scale, br);
      else if (Ci == 64)
        attn_k<64><<<g, 32, 0, stream>>>(Qb, Kbuf[br], Vtb[br], Ob, Ni, Ns[j], scale, br);
      else
        attn_k<128><<<g, 32, 0, stream>>>(Qb, Kbuf[br], Vtb[br], Ob, Ni, Ns[j], scale, br);
    }
    // 6) Output projection -> f32, scattered straight into NCHW d_out
    //    Ncols = Ci: 32 -> NT=2 (one tile-col), 64 -> NT=4, 128 -> NT=4 x 2 blocks
    if (Ci == 32) {
      gemm_k<2, 2><<<dim3((BATCH * Ni) / 16, 1), 32, 0, stream>>>(
          Ob, woT, 2 * inner, Ci, nullptr, nullptr, outf + outOff[i], 0, 0, Ni, Ci);
    } else {
      gemm_k<4, 2><<<dim3((BATCH * Ni) / 16, Ci / 64), 32, 0, stream>>>(
          Ob, woT, 2 * inner, Ci, nullptr, nullptr, outf + outOff[i], 0, 0, Ni, Ci);
    }
  }
}
